// AdaUp_65841848648171
// MI455X (gfx1250) — matrix operand order repaired and checked by reading, not compiled, not run on hardware
//
#include <hip/hip_runtime.h>
#include <hip/hip_bf16.h>

typedef __attribute__((ext_vector_type(16))) _Float16 v16h;
typedef __attribute__((ext_vector_type(8)))  _Float16 v8h;
typedef __attribute__((ext_vector_type(8)))  float    v8f;

#define BATCH 8
#define CIN   256
#define CTOT  512
#define COUT  256
#define HS    64
#define WS    64
#define HU    128
#define WU    128
#define NPIX_UP (HU*WU)      // 16384
#define NPIX    (HS*WS)      // 4096
#define NPTS  4

// ---------------------------------------------------------------------------
// f32 -> f16 weight conversion
// ---------------------------------------------------------------------------
__global__ void cvt_f16_kernel(const float* __restrict__ in,
                               _Float16* __restrict__ out, int n) {
    int i = blockIdx.x * 256 + threadIdx.x;
    if (i < n) out[i] = (_Float16)in[i];
}

// ---------------------------------------------------------------------------
// conv1: off_feat = relu(W1 @ concat(upsample(high), low) + b1)
// GEMM: M=256 (out ch), K=512, N=16384 per batch.  One workgroup = full M x
// 64-pixel N tile; upsample of the high half is fused into the LDS fill.
// 8 waves, each wave: 32 M-rows x 64 N-cols = 2x4 WMMA 16x16 tiles.
// ---------------------------------------------------------------------------
__global__ __launch_bounds__(256) void gemm_conv1_kernel(
    const float* __restrict__ high, const float* __restrict__ low,
    const _Float16* __restrict__ W1h, const float* __restrict__ b1,
    _Float16* __restrict__ offF)
{
    __shared__ __align__(16) _Float16 sB[64 * 40];   // [n_local][k] stride 40

    const int b    = blockIdx.y;
    const int tile = blockIdx.x;                 // 0..255
    const int t    = threadIdx.x;
    const int lane = t & 31;
    const int wave = t >> 5;
    const int l15  = lane & 15;
    const int half = lane >> 4;                  // 0 or 1
    const int pixBase = tile * 64;

    // ---- fill-role mapping: thread fills column nf, k rows kq*8..kq*8+7 ----
    const int nf  = t & 63;
    const int kq  = t >> 6;                      // 0..3
    const int pix = pixBase + nf;
    const int py  = pix >> 7;                    // y in [0,128)
    const int px  = pix & 127;                   // x in [0,128)

    // bilinear (align_corners=false) source coords on 64x64, edge-clamped
    const float sxf = 0.5f * (float)px - 0.25f;
    const float syf = 0.5f * (float)py - 0.25f;
    const float x0f = floorf(sxf), y0f = floorf(syf);
    const float fx = sxf - x0f,   fy = syf - y0f;
    int ix0 = (int)x0f, iy0 = (int)y0f;
    int ix1 = min(max(ix0 + 1, 0), WS - 1);
    int iy1 = min(max(iy0 + 1, 0), HS - 1);
    ix0 = min(max(ix0, 0), WS - 1);
    iy0 = min(max(iy0, 0), HS - 1);
    const int i00 = iy0 * WS + ix0, i01 = iy0 * WS + ix1;
    const int i10 = iy1 * WS + ix0, i11 = iy1 * WS + ix1;
    const float w00 = (1.f - fx) * (1.f - fy), w01 = fx * (1.f - fy);
    const float w10 = (1.f - fx) * fy,         w11 = fx * fy;

    v8f acc[2][4];
#pragma unroll
    for (int mt = 0; mt < 2; ++mt)
#pragma unroll
        for (int nt = 0; nt < 4; ++nt) acc[mt][nt] = (v8f){0,0,0,0,0,0,0,0};

    const int koff = half * 16;                  // A-frag K sub-offset

    for (int K0 = 0; K0 < CTOT; K0 += 32) {
        __syncthreads();
        // -------- stage B tile [64 pixels][32 k] as f16 into LDS --------
#pragma unroll
        for (int j = 0; j < 8; ++j) {
            const int kl = kq * 8 + j;
            const int k  = K0 + kl;
            float v;
            if (k < CIN) {
                const float* hp = high + ((size_t)(b * CIN + k)) * NPIX;
                v = w00 * hp[i00] + w01 * hp[i01] + w10 * hp[i10] + w11 * hp[i11];
            } else {
                v = low[((size_t)(b * COUT + (k - CIN))) * NPIX_UP + pix];
            }
            sB[nf * 40 + kl] = (_Float16)v;
        }
        if (K0 + 32 >= CIN && K0 + 32 < CTOT)    // gfx1250 global_prefetch_b8
            __builtin_prefetch(low + ((size_t)(b * COUT + (K0 + 32 - CIN))) * NPIX_UP + pix, 0, 0);
        __syncthreads();

        // -------- B fragments from LDS (K-contiguous per lane) --------
        v16h bf[4];
#pragma unroll
        for (int nt = 0; nt < 4; ++nt) {
            const _Float16* bp = &sB[(nt * 16 + l15) * 40 + half * 16];
            v8h b0 = *(const v8h*)bp;
            v8h b1v = *(const v8h*)(bp + 8);
#pragma unroll
            for (int i = 0; i < 8; ++i) { bf[nt][i] = b0[i]; bf[nt][i + 8] = b1v[i]; }
        }
        // -------- A fragments from global f16 weights + WMMA --------
#pragma unroll
        for (int mt = 0; mt < 2; ++mt) {
            const int m = wave * 32 + mt * 16 + l15;
            const _Float16* ap = W1h + (size_t)m * CTOT + K0 + koff;
            v8h a0 = *(const v8h*)ap;
            v8h a1 = *(const v8h*)(ap + 8);
            v16h a;
#pragma unroll
            for (int i = 0; i < 8; ++i) { a[i] = a0[i]; a[i + 8] = a1[i]; }
#pragma unroll
            for (int nt = 0; nt < 4; ++nt)
                acc[mt][nt] = __builtin_amdgcn_wmma_f32_16x16x32_f16(
                    false, a, false, bf[nt], (short)0, acc[mt][nt], false, false);
        }
    }

    // -------- epilogue: +bias, relu, f16 store --------
#pragma unroll
    for (int mt = 0; mt < 2; ++mt)
#pragma unroll
        for (int nt = 0; nt < 4; ++nt)
#pragma unroll
            for (int i = 0; i < 8; ++i) {
                const int m = wave * 32 + mt * 16 + half * 8 + i;
                const int n = pixBase + nt * 16 + l15;
                float v = acc[mt][nt][i] + b1[m];
                offF[((size_t)(b * CIN + m)) * NPIX_UP + n] = (_Float16)fmaxf(v, 0.f);
            }
}

// ---------------------------------------------------------------------------
// convf: proc = Wf @ high + bf   (M=256, K=256, N=4096 per batch)
// ---------------------------------------------------------------------------
__global__ __launch_bounds__(256) void gemm_convf_kernel(
    const float* __restrict__ high, const _Float16* __restrict__ Wfh,
    const float* __restrict__ bf_, float* __restrict__ proc)
{
    __shared__ __align__(16) _Float16 sB[64 * 40];

    const int b    = blockIdx.y;
    const int tile = blockIdx.x;                 // 0..63
    const int t    = threadIdx.x;
    const int lane = t & 31;
    const int wave = t >> 5;
    const int l15  = lane & 15;
    const int half = lane >> 4;
    const int pixBase = tile * 64;

    const int nf  = t & 63;
    const int kq  = t >> 6;
    const int pix = pixBase + nf;

    v8f acc[2][4];
#pragma unroll
    for (int mt = 0; mt < 2; ++mt)
#pragma unroll
        for (int nt = 0; nt < 4; ++nt) acc[mt][nt] = (v8f){0,0,0,0,0,0,0,0};

    const int koff = half * 16;

    for (int K0 = 0; K0 < CIN; K0 += 32) {
        __syncthreads();
#pragma unroll
        for (int j = 0; j < 8; ++j) {
            const int kl = kq * 8 + j;
            const int k  = K0 + kl;
            sB[nf * 40 + kl] =
                (_Float16)high[((size_t)(b * CIN + k)) * NPIX + pix];
        }
        __syncthreads();

        v16h bf[4];
#pragma unroll
        for (int nt = 0; nt < 4; ++nt) {
            const _Float16* bp = &sB[(nt * 16 + l15) * 40 + half * 16];
            v8h b0 = *(const v8h*)bp;
            v8h b1v = *(const v8h*)(bp + 8);
#pragma unroll
            for (int i = 0; i < 8; ++i) { bf[nt][i] = b0[i]; bf[nt][i + 8] = b1v[i]; }
        }
#pragma unroll
        for (int mt = 0; mt < 2; ++mt) {
            const int m = wave * 32 + mt * 16 + l15;
            const _Float16* ap = Wfh + (size_t)m * CIN + K0 + koff;
            v8h a0 = *(const v8h*)ap;
            v8h a1 = *(const v8h*)(ap + 8);
            v16h a;
#pragma unroll
            for (int i = 0; i < 8; ++i) { a[i] = a0[i]; a[i + 8] = a1[i]; }
#pragma unroll
            for (int nt = 0; nt < 4; ++nt)
                acc[mt][nt] = __builtin_amdgcn_wmma_f32_16x16x32_f16(
                    false, a, false, bf[nt], (short)0, acc[mt][nt], false, false);
        }
    }

#pragma unroll
    for (int mt = 0; mt < 2; ++mt)
#pragma unroll
        for (int nt = 0; nt < 4; ++nt)
#pragma unroll
            for (int i = 0; i < 8; ++i) {
                const int m = wave * 32 + mt * 16 + half * 8 + i;
                const int n = pixBase + nt * 16 + l15;
                proc[((size_t)(b * COUT + m)) * NPIX + n] = acc[mt][nt][i] + bf_[m];
            }
}

// ---------------------------------------------------------------------------
// conv3x3 (pad=1, 256 -> 8 channels): offsets = W2 * off_feat + b2
// LDS-tiled direct conv, 16x16 pixel tile, 8-channel input chunks.
// ---------------------------------------------------------------------------
__global__ __launch_bounds__(256) void conv3x3_kernel(
    const _Float16* __restrict__ offF, const float* __restrict__ W2,
    const float* __restrict__ b2, float* __restrict__ offs)
{
    __shared__ _Float16 sP[8][18][20];
    __shared__ float    sW[8][8][9];            // [c_local][o][tap]

    const int b  = blockIdx.z;
    const int ty = blockIdx.y, tx = blockIdx.x;
    const int t  = threadIdx.x;
    const int py = t >> 4, px = t & 15;
    const int gy = ty * 16 + py, gx = tx * 16 + px;

    float acc[8];
#pragma unroll
    for (int o = 0; o < 8; ++o) acc[o] = 0.f;

    for (int cc = 0; cc < CIN; cc += 8) {
        __syncthreads();
        for (int idx = t; idx < 8 * 18 * 18; idx += 256) {
            const int c  = idx / 324;
            const int r  = idx - c * 324;
            const int yy = r / 18, xx = r - (r / 18) * 18;
            const int sy = ty * 16 + yy - 1;
            const int sx = tx * 16 + xx - 1;
            _Float16 v = (_Float16)0.f;
            if (sy >= 0 && sy < HU && sx >= 0 && sx < WU)
                v = offF[((size_t)(b * CIN + cc + c)) * NPIX_UP + sy * WU + sx];
            sP[c][yy][xx] = v;
        }
        for (int idx = t; idx < 576; idx += 256) {
            const int o  = idx / 72;
            const int r  = idx - o * 72;
            const int c  = r / 9;
            const int kk = r - c * 9;
            sW[c][o][kk] = W2[((size_t)(o * CIN + cc + c)) * 9 + kk];
        }
        __syncthreads();
#pragma unroll
        for (int c = 0; c < 8; ++c) {
            float v[9];
#pragma unroll
            for (int ky = 0; ky < 3; ++ky)
#pragma unroll
                for (int kx = 0; kx < 3; ++kx)
                    v[ky * 3 + kx] = (float)sP[c][py + ky][px + kx];
#pragma unroll
            for (int o = 0; o < 8; ++o) {
                float s = acc[o];
#pragma unroll
                for (int kk = 0; kk < 9; ++kk) s = fmaf(sW[c][o][kk], v[kk], s);
                acc[o] = s;
            }
        }
    }
#pragma unroll
    for (int o = 0; o < 8; ++o)
        offs[((size_t)(b * 8 + o)) * NPIX_UP + gy * WU + gx] = acc[o] + b2[o];
}

// ---------------------------------------------------------------------------
// Deformable sampling: out = mean_p grid_sample_border(proc, base + off_p)
// One block per output row; indices/weights computed once, 256-ch stream.
// ---------------------------------------------------------------------------
__global__ __launch_bounds__(128) void sample_avg_kernel(
    const float* __restrict__ proc, const float* __restrict__ offs,
    float* __restrict__ out)
{
    const int b = blockIdx.y;
    const int y = blockIdx.x;
    const int x = threadIdx.x;
    const int pix = y * WU + x;

    const float gxb = -1.f + 2.f * (float)x * (1.f / (WU - 1));
    const float gyb = -1.f + 2.f * (float)y * (1.f / (HU - 1));

    int   idx[NPTS][4];
    float wgt[NPTS][4];
#pragma unroll
    for (int p = 0; p < NPTS; ++p) {
        const float ox = offs[((size_t)(b * 8 + 2 * p))     * NPIX_UP + pix];
        const float oy = offs[((size_t)(b * 8 + 2 * p + 1)) * NPIX_UP + pix];
        const float gx = gxb + ox * (2.f / WU);
        const float gy = gyb + oy * (2.f / HU);
        float pxf = fminf(fmaxf((gx + 1.f) * (WS * 0.5f) - 0.5f, 0.f), (float)(WS - 1));
        float pyf = fminf(fmaxf((gy + 1.f) * (HS * 0.5f) - 0.5f, 0.f), (float)(HS - 1));
        const float x0f = floorf(pxf), y0f = floorf(pyf);
        const float fx = pxf - x0f,   fy = pyf - y0f;
        const int x0 = (int)x0f, y0 = (int)y0f;
        const int x1 = min(x0 + 1, WS - 1), y1 = min(y0 + 1, HS - 1);
        idx[p][0] = y0 * WS + x0; idx[p][1] = y0 * WS + x1;
        idx[p][2] = y1 * WS + x0; idx[p][3] = y1 * WS + x1;
        wgt[p][0] = (1.f - fx) * (1.f - fy); wgt[p][1] = fx * (1.f - fy);
        wgt[p][2] = (1.f - fx) * fy;         wgt[p][3] = fx * fy;
    }

    for (int c = 0; c < COUT; ++c) {
        const float* pp = proc + ((size_t)(b * COUT + c)) * NPIX;
        float s = 0.f;
#pragma unroll
        for (int p = 0; p < NPTS; ++p)
            s += wgt[p][0] * pp[idx[p][0]] + wgt[p][1] * pp[idx[p][1]]
               + wgt[p][2] * pp[idx[p][2]] + wgt[p][3] * pp[idx[p][3]];
        out[((size_t)(b * COUT + c)) * NPIX_UP + pix] = 0.25f * s;
    }
}

// ---------------------------------------------------------------------------
extern "C" void kernel_launch(void* const* d_in, const int* in_sizes, int n_in,
                              void* d_out, int out_size, void* d_ws, size_t ws_size,
                              hipStream_t stream) {
    const float* high = (const float*)d_in[0];
    const float* low  = (const float*)d_in[1];
    const float* W1   = (const float*)d_in[2];
    const float* b1   = (const float*)d_in[3];
    const float* W2   = (const float*)d_in[4];
    const float* b2   = (const float*)d_in[5];
    const float* Wf   = (const float*)d_in[6];
    const float* bfp  = (const float*)d_in[7];

    char* ws = (char*)d_ws;
    size_t o = 0;
    _Float16* W1h = (_Float16*)(ws + o); o += (size_t)CIN * CTOT * 2;        // 256 KiB
    _Float16* Wfh = (_Float16*)(ws + o); o += (size_t)COUT * CIN * 2;        // 128 KiB
    _Float16* offF = (_Float16*)(ws + o); o += (size_t)BATCH * CIN * NPIX_UP * 2;   // 64 MiB
    float* procB = (float*)(ws + o);     o += (size_t)BATCH * COUT * NPIX * 4;      // 32 MiB
    float* offsB = (float*)(ws + o);     o += (size_t)BATCH * 8 * NPIX_UP * 4;      // 4 MiB
    (void)ws_size; (void)in_sizes; (void)n_in; (void)out_size;

    cvt_f16_kernel<<<(CIN * CTOT) / 256, 256, 0, stream>>>(W1, W1h, CIN * CTOT);
    cvt_f16_kernel<<<(COUT * CIN) / 256, 256, 0, stream>>>(Wf, Wfh, COUT * CIN);

    gemm_conv1_kernel<<<dim3(NPIX_UP / 64, BATCH), 256, 0, stream>>>(
        high, low, W1h, b1, offF);
    gemm_convf_kernel<<<dim3(NPIX / 64, BATCH), 256, 0, stream>>>(
        high, Wfh, bfp, procB);
    conv3x3_kernel<<<dim3(WU / 16, HU / 16, BATCH), 256, 0, stream>>>(
        offF, W2, b2, offsB);
    sample_avg_kernel<<<dim3(HU, BATCH), WU, 0, stream>>>(
        procB, offsB, (float*)d_out);
}
